// RNN_44092134260747
// MI455X (gfx1250) — matrix operand order repaired and checked by reading, not compiled, not run on hardware
//
#include <hip/hip_runtime.h>
#include <hip/hip_fp16.h>
#include <math.h>

typedef __attribute__((ext_vector_type(16))) _Float16 v16h;
typedef __attribute__((ext_vector_type(8)))  _Float16 v8h;
typedef __attribute__((ext_vector_type(8)))  float    v8f;

#define NB 128
#define TT 512
#define DD 1024
#define HH 1024
#define NBLK 64   // persistent-kernel blocks (2 waves each -> 128 wave tasks)

__device__ __forceinline__ v8f wmma16(v16h a, v16h b, v8f c) {
  return __builtin_amdgcn_wmma_f32_16x16x32_f16(false, a, false, b, (short)0, c, false, false);
}
__device__ __forceinline__ v8f wmma16_ra(v16h a, v16h b, v8f c) {
  return __builtin_amdgcn_wmma_f32_16x16x32_f16(false, a, false, b, (short)0, c, true, false);
}

__device__ __forceinline__ v16h cvt_a(v8f a0, v8f a1) {
  v16h a;
#pragma unroll
  for (int e = 0; e < 8; ++e) { a[e] = (_Float16)a0[e]; a[e + 8] = (_Float16)a1[e]; }
  return a;
}
__device__ __forceinline__ v16h cat_a(v8h a0, v8h a1) {
  return __builtin_shufflevector(a0, a1, 0, 1, 2, 3, 4, 5, 6, 7,
                                 8, 9, 10, 11, 12, 13, 14, 15);
}
__device__ __forceinline__ void mm4(v16h a, v16h b0, v16h b1, v16h b2, v16h b3,
                                    v8f acc[4]) {
  acc[0] = wmma16(a, b0, acc[0]);
  acc[1] = wmma16_ra(a, b1, acc[1]);
  acc[2] = wmma16_ra(a, b2, acc[2]);
  acc[3] = wmma16_ra(a, b3, acc[3]);
}

// fp32 [K][N] row-major -> f16 [N][K] row-major (transpose + convert)
__global__ void cvt_transpose_kernel(const float* __restrict__ in,
                                     _Float16* __restrict__ out, int K, int N) {
  int idx = blockIdx.x * blockDim.x + threadIdx.x;
  if (idx >= K * N) return;
  int k = idx / N;
  int n = idx - k * N;
  out[(size_t)n * K + k] = (_Float16)in[idx];
}

__global__ void cvt_h0_kernel(const float* __restrict__ h0,
                              _Float16* __restrict__ hbuf, unsigned* bar) {
  int idx = blockIdx.x * blockDim.x + threadIdx.x;  // < NB*HH
  hbuf[idx] = (_Float16)h0[idx];
  if (idx == 0) *bar = 0u;
}

// Phase 1: out[m][n] = sum_k x[m][k]*Wx[k][n] + b[n].
// One wave -> 16x64 tile; unroll-2 ping-pong pipelined K loop (no reg rotation).
__global__ void __launch_bounds__(256)
gemm_xw_kernel(const float* __restrict__ x, const _Float16* __restrict__ WxT,
               const float* __restrict__ bias, float* __restrict__ out) {
  int lane = threadIdx.x & 31;
  int wave = threadIdx.x >> 5;
  int task = blockIdx.x * 8 + wave;      // 0..65535
  int nStrip = task & 15;
  int mTile  = task >> 4;                // 0..4095
  int mBase = mTile << 4;
  int nBase = nStrip << 6;
  int l15 = lane & 15;
  int hi  = lane >> 4;

  const float*    Arow  = x + (size_t)(mBase + l15) * DD + hi * 16;
  const _Float16* Bbase = WxT + (size_t)(nBase + l15) * DD + hi * 16;

  v8f acc[4] = {v8f{}, v8f{}, v8f{}, v8f{}};

  // set0 @ k=0
  v8f  a0 = *(const v8f*)(Arow);
  v8f  a1 = *(const v8f*)(Arow + 8);
  v16h b0 = *(const v16h*)(Bbase);
  v16h b1 = *(const v16h*)(Bbase + 16 * DD);
  v16h b2 = *(const v16h*)(Bbase + 32 * DD);
  v16h b3 = *(const v16h*)(Bbase + 48 * DD);

  for (int k0 = 0; k0 < DD - 64; k0 += 64) {
    // set1 @ k0+32 (overlaps compute of set0)
    v8f  a0x = *(const v8f*)(Arow + k0 + 32);
    v8f  a1x = *(const v8f*)(Arow + k0 + 40);
    v16h c0 = *(const v16h*)(Bbase + k0 + 32);
    v16h c1 = *(const v16h*)(Bbase + k0 + 32 + 16 * DD);
    v16h c2 = *(const v16h*)(Bbase + k0 + 32 + 32 * DD);
    v16h c3 = *(const v16h*)(Bbase + k0 + 32 + 48 * DD);
    mm4(cvt_a(a0, a1), b0, b1, b2, b3, acc);

    // set0 @ k0+64 (overlaps compute of set1); redefined by loads, no copies
    a0 = *(const v8f*)(Arow + k0 + 64);
    a1 = *(const v8f*)(Arow + k0 + 72);
    b0 = *(const v16h*)(Bbase + k0 + 64);
    b1 = *(const v16h*)(Bbase + k0 + 64 + 16 * DD);
    b2 = *(const v16h*)(Bbase + k0 + 64 + 32 * DD);
    b3 = *(const v16h*)(Bbase + k0 + 64 + 48 * DD);
    mm4(cvt_a(a0x, a1x), c0, c1, c2, c3, acc);
  }
  {  // epilogue: k = DD-64 (in set0) and k = DD-32
    v8f  a0x = *(const v8f*)(Arow + DD - 32);
    v8f  a1x = *(const v8f*)(Arow + DD - 24);
    v16h c0 = *(const v16h*)(Bbase + DD - 32);
    v16h c1 = *(const v16h*)(Bbase + DD - 32 + 16 * DD);
    v16h c2 = *(const v16h*)(Bbase + DD - 32 + 32 * DD);
    v16h c3 = *(const v16h*)(Bbase + DD - 32 + 48 * DD);
    mm4(cvt_a(a0, a1), b0, b1, b2, b3, acc);
    mm4(cvt_a(a0x, a1x), c0, c1, c2, c3, acc);
  }

  int rowHi = hi * 8;
#pragma unroll
  for (int j = 0; j < 4; ++j) {
    int col = nBase + j * 16 + l15;
    float bv = bias[col];
#pragma unroll
    for (int r = 0; r < 8; ++r) {
      out[(size_t)(mBase + rowHi + r) * HH + col] = acc[j][r] + bv;
    }
  }
}

// Phase 2: persistent kernel, all 512 timesteps with an agent-scope grid
// barrier between steps (removes 511 launch latencies from the scan's
// critical path). 64 blocks x 2 waves = 128 wave tasks = 8 Mtiles x 16 Nstrips.
__global__ void __launch_bounds__(64)
rnn_scan_kernel(_Float16* __restrict__ hA, _Float16* __restrict__ hB,
                const _Float16* __restrict__ WhT, float* __restrict__ out,
                unsigned* bar) {
  int lane = threadIdx.x & 31;
  int wave = threadIdx.x >> 5;
  int task = blockIdx.x * 2 + wave;      // 0..127
  int nStrip = task & 15;
  int mTile  = task >> 4;                // 0..7
  int mBase = mTile << 4;
  int nBase = nStrip << 6;
  int l15 = lane & 15;
  int hi  = lane >> 4;
  int rowHi = hi * 8;

  const _Float16* Bbase = WhT + (size_t)(nBase + l15) * HH + hi * 16;
  const size_t arowOff = (size_t)(mBase + l15) * HH + hi * 16;

  for (int t = 0; t < TT; ++t) {
    const _Float16* hin  = (t & 1) ? hB : hA;
    _Float16*       hout = (t & 1) ? hA : hB;
    const _Float16* Arow = hin + arowOff;

    // Pull next timestep's xw slice toward L2 while we compute this one.
    if (t + 1 < TT) {
      const float* pf =
          out + ((size_t)(mBase + l15) * TT + (t + 1)) * HH + nBase + hi * 32;
      __builtin_prefetch(pf, 0, 1);
    }

    v8f acc[4] = {v8f{}, v8f{}, v8f{}, v8f{}};

    v8h  a0 = *(const v8h*)(Arow);
    v8h  a1 = *(const v8h*)(Arow + 8);
    v16h b0 = *(const v16h*)(Bbase);
    v16h b1 = *(const v16h*)(Bbase + 16 * HH);
    v16h b2 = *(const v16h*)(Bbase + 32 * HH);
    v16h b3 = *(const v16h*)(Bbase + 48 * HH);

    for (int k0 = 0; k0 < HH - 64; k0 += 64) {
      v8h  a0x = *(const v8h*)(Arow + k0 + 32);
      v8h  a1x = *(const v8h*)(Arow + k0 + 40);
      v16h c0 = *(const v16h*)(Bbase + k0 + 32);
      v16h c1 = *(const v16h*)(Bbase + k0 + 32 + 16 * HH);
      v16h c2 = *(const v16h*)(Bbase + k0 + 32 + 32 * HH);
      v16h c3 = *(const v16h*)(Bbase + k0 + 32 + 48 * HH);
      mm4(cat_a(a0, a1), b0, b1, b2, b3, acc);

      a0 = *(const v8h*)(Arow + k0 + 64);
      a1 = *(const v8h*)(Arow + k0 + 72);
      b0 = *(const v16h*)(Bbase + k0 + 64);
      b1 = *(const v16h*)(Bbase + k0 + 64 + 16 * HH);
      b2 = *(const v16h*)(Bbase + k0 + 64 + 32 * HH);
      b3 = *(const v16h*)(Bbase + k0 + 64 + 48 * HH);
      mm4(cat_a(a0x, a1x), c0, c1, c2, c3, acc);
    }
    {
      v8h  a0x = *(const v8h*)(Arow + HH - 32);
      v8h  a1x = *(const v8h*)(Arow + HH - 24);
      v16h c0 = *(const v16h*)(Bbase + HH - 32);
      v16h c1 = *(const v16h*)(Bbase + HH - 32 + 16 * HH);
      v16h c2 = *(const v16h*)(Bbase + HH - 32 + 32 * HH);
      v16h c3 = *(const v16h*)(Bbase + HH - 32 + 48 * HH);
      mm4(cat_a(a0, a1), b0, b1, b2, b3, acc);
      mm4(cat_a(a0x, a1x), c0, c1, c2, c3, acc);
    }

#pragma unroll
    for (int j = 0; j < 4; ++j) {
      int col = nBase + j * 16 + l15;
#pragma unroll
      for (int r = 0; r < 8; ++r) {
        int m = mBase + rowHi + r;                      // batch index
        size_t oidx = ((size_t)m * TT + t) * HH + col;  // (m, t, col)
        float v = tanhf(acc[j][r] + out[oidx]);         // xw_t read in place
        out[oidx] = v;                                  // overwritten with h_t
        hout[(size_t)m * HH + col] = (_Float16)v;
      }
    }

    // ---- grid-wide barrier (monotonic counter, no reset races) ----
    __syncthreads();
    if (threadIdx.x == 0) {
      __hip_atomic_fetch_add(bar, 1u, __ATOMIC_RELEASE, __HIP_MEMORY_SCOPE_AGENT);
      unsigned target = (unsigned)(t + 1) * (unsigned)NBLK;
      while (__hip_atomic_load(bar, __ATOMIC_ACQUIRE, __HIP_MEMORY_SCOPE_AGENT) < target)
        __builtin_amdgcn_s_sleep(2);
    }
    __syncthreads();
  }
}

extern "C" void kernel_launch(void* const* d_in, const int* in_sizes, int n_in,
                              void* d_out, int out_size, void* d_ws, size_t ws_size,
                              hipStream_t stream) {
  const float* x  = (const float*)d_in[0];   // (N,T,D)
  const float* h0 = (const float*)d_in[1];   // (N,H)
  const float* Wx = (const float*)d_in[2];   // (D,H)
  const float* Wh = (const float*)d_in[3];   // (H,H)
  const float* b  = (const float*)d_in[4];   // (H,)
  float* out = (float*)d_out;                // (N,T,H)

  char* ws = (char*)d_ws;
  _Float16* WxT = (_Float16*)ws;                                   // 2 MB
  _Float16* WhT = (_Float16*)(ws + (size_t)DD * HH * 2);           // 2 MB
  _Float16* hA  = (_Float16*)(ws + (size_t)DD * HH * 4);           // 256 KB
  _Float16* hB  = hA + (size_t)NB * HH;                            // 256 KB
  unsigned* bar = (unsigned*)(hB + (size_t)NB * HH);               // 4 B

  cvt_transpose_kernel<<<(DD * HH + 255) / 256, 256, 0, stream>>>(Wx, WxT, DD, HH);
  cvt_transpose_kernel<<<(HH * HH + 255) / 256, 256, 0, stream>>>(Wh, WhT, HH, HH);
  cvt_h0_kernel<<<(NB * HH) / 256, 256, 0, stream>>>(h0, hA, bar);

  gemm_xw_kernel<<<8192, 256, 0, stream>>>(x, WxT, b, out);

  rnn_scan_kernel<<<NBLK, 64, 0, stream>>>(hA, hB, WhT, out, bar);
}
